// Qwen3Attention_71365176590387
// MI455X (gfx1250) — matrix operand order repaired and checked by reading, not compiled, not run on hardware
//
#include <hip/hip_runtime.h>
#include <stdint.h>
#include <stddef.h>

// ---------------- types ----------------
typedef __attribute__((ext_vector_type(16))) __bf16 v16bf;
typedef __attribute__((ext_vector_type(2)))  __bf16 v2bf;
typedef __attribute__((ext_vector_type(8)))  float  v8f;
typedef __attribute__((ext_vector_type(4)))  unsigned int u32x4;
typedef __attribute__((ext_vector_type(8)))  int    i32x8;
typedef __attribute__((ext_vector_type(4)))  int    i32x4;

#define T_SEQ     2048
#define HIDDEN    4096
#define NUM_HEADS 32
#define NUM_KV    8
#define HEAD_DIM  128
#define QKV_OUT   6144   // (32 + 16) * 128
#define Q_SIZE    4096
#define KV_SIZE   1024
#define ATTN_SCALE 0.08838834764831845f  // 128^-0.5

static __device__ __forceinline__ unsigned short f2bf(float f) {
    union { float f; unsigned u; } c; c.f = f;
    unsigned u = c.u;
    unsigned r = u + 0x7fffu + ((u >> 16) & 1u);   // round-to-nearest-even
    return (unsigned short)(r >> 16);
}

// packed f32x2 -> bf16x2 (one v_cvt_pk_bf16_f32 if the builtin exists)
static __device__ __forceinline__ unsigned pk2bf(float lo, float hi) {
#if __has_builtin(__builtin_amdgcn_cvt_pk_bf16_f32)
    union { v2bf v; unsigned u; } c;
    c.v = __builtin_amdgcn_cvt_pk_bf16_f32(lo, hi);
    return c.u;
#else
    return (unsigned)f2bf(lo) | ((unsigned)f2bf(hi) << 16);
#endif
}

static __device__ __forceinline__ v8f vzero8() {
    v8f v = {0.f,0.f,0.f,0.f,0.f,0.f,0.f,0.f};
    return v;
}

// ---------------- DPP16 16-lane butterfly reductions (no LDS traffic) ----
// xor1,xor2 via quad_perm; then cross-quad via row_half_mirror (quads are
// uniform after the first two stages) and cross-8 via row_mirror.
#define DPP_QP_XOR1 0xB1   // quad_perm [1,0,3,2]
#define DPP_QP_XOR2 0x4E   // quad_perm [2,3,0,1]
#define DPP_RHMIRR  0x141  // row_half_mirror
#define DPP_RMIRR   0x140  // row_mirror

static __device__ __forceinline__ float dpp_max16(float x) {
    int t;
    t = __builtin_amdgcn_mov_dpp(__float_as_int(x), DPP_QP_XOR1, 0xF, 0xF, false);
    x = fmaxf(x, __int_as_float(t));
    t = __builtin_amdgcn_mov_dpp(__float_as_int(x), DPP_QP_XOR2, 0xF, 0xF, false);
    x = fmaxf(x, __int_as_float(t));
    t = __builtin_amdgcn_mov_dpp(__float_as_int(x), DPP_RHMIRR, 0xF, 0xF, false);
    x = fmaxf(x, __int_as_float(t));
    t = __builtin_amdgcn_mov_dpp(__float_as_int(x), DPP_RMIRR, 0xF, 0xF, false);
    x = fmaxf(x, __int_as_float(t));
    return x;
}

static __device__ __forceinline__ float dpp_add16(float x) {
    int t;
    t = __builtin_amdgcn_mov_dpp(__float_as_int(x), DPP_QP_XOR1, 0xF, 0xF, false);
    x += __int_as_float(t);
    t = __builtin_amdgcn_mov_dpp(__float_as_int(x), DPP_QP_XOR2, 0xF, 0xF, false);
    x += __int_as_float(t);
    t = __builtin_amdgcn_mov_dpp(__float_as_int(x), DPP_RHMIRR, 0xF, 0xF, false);
    x += __int_as_float(t);
    t = __builtin_amdgcn_mov_dpp(__float_as_int(x), DPP_RMIRR, 0xF, 0xF, false);
    x += __int_as_float(t);
    return x;
}

// ---------------- TDM 2D tile load: global(bf16) -> LDS ----------------
// Descriptor per CDNA5 ISA sec 8 (groups 0/1; 2D tensor -> groups 2/3 zero).
static __device__ __forceinline__ void tdm_load_tile_2d(
    unsigned lds_off_bytes, const void* gptr,
    unsigned tile_rows, unsigned row_elems, unsigned row_stride_elems)
{
    unsigned long long ga = (unsigned long long)(uintptr_t)gptr;
    u32x4 g0;
    g0[0] = 1u;                                      // count=1, user mode
    g0[1] = lds_off_bytes;                           // lds_addr
    g0[2] = (unsigned)(ga & 0xffffffffull);          // global_addr[31:0]
    g0[3] = (unsigned)((ga >> 32) & 0x1ffffffull) |  // global_addr[56:32]
            (2u << 30);                              // type=2 ("image")
    i32x8 g1;
    g1[0] = (int)(1u << 16);                         // workgroup_mask=0, data_size=1 (2B)
    g1[1] = (int)((row_elems & 0xffffu) << 16);      // tensor_dim0[15:0]
    g1[2] = (int)(row_elems >> 16);                  // tensor_dim0 hi | tensor_dim1 lo16 (=0)
    g1[3] = (int)(16u | ((row_elems & 0xffffu) << 16)); // tensor_dim1 hi (1<<20), tile_dim0
    g1[4] = (int)(tile_rows & 0xffffu);              // tile_dim1 (tile_dim2=0)
    g1[5] = (int)row_stride_elems;                   // tensor_dim0_stride[31:0]
    g1[6] = 0;                                       // stride hi / dim1_stride lo
    g1[7] = 0;
    i32x4 g2; g2[0]=0; g2[1]=0; g2[2]=0; g2[3]=0;
    i32x4 g3; g3[0]=0; g3[1]=0; g3[2]=0; g3[3]=0;
#if defined(__clang_major__) && (__clang_major__ >= 23)
    i32x8 g4; g4[0]=0; g4[1]=0; g4[2]=0; g4[3]=0; g4[4]=0; g4[5]=0; g4[6]=0; g4[7]=0;
    __builtin_amdgcn_tensor_load_to_lds(g0, g1, g2, g3, g4, 0);
#else
    __builtin_amdgcn_tensor_load_to_lds(g0, g1, g2, g3, 0);
#endif
}

// fragment load helpers: two 16B LDS reads -> v16bf
static __device__ __forceinline__ v16bf frag2(const unsigned short* p0,
                                              const unsigned short* p1) {
    v16bf f;
    ((int4*)&f)[0] = *(const int4*)p0;
    ((int4*)&f)[1] = *(const int4*)p1;
    return f;
}

// ======================================================================
// Kernel 1/4: tiled GEMM with bf16 WMMA.  C(f32)[M,N] = A[M,K] * B(f32)[K,N]
// A is fp32 (ABF16=false) or bf16 (ABF16=true); both converted/staged in LDS.
// Workgroup: 256 thr = 8 waves, tile 128x128; wave tile 32x64 (2x4 WMMA accs).
// ======================================================================
template <bool ABF16>
__global__ __launch_bounds__(256) void gemm_bf16_wmma(
    const void* __restrict__ Av, const float* __restrict__ B,
    float* __restrict__ C, int M, int N, int K)
{
    __shared__ unsigned short As[128][48];   // [m][k] row-major, K<=32 used
    __shared__ unsigned short Bs[128][48];   // [n][k] (B staged transposed)

    const int tid  = threadIdx.x;
    const int lane = tid & 31;
    const int w    = tid >> 5;
    const int m0   = blockIdx.y * 128;
    const int n0   = blockIdx.x * 128;
    const int wm   = (w & 3) * 32;
    const int wn   = (w >> 2) * 64;

    v8f acc[2][4];
#pragma unroll
    for (int mi = 0; mi < 2; ++mi)
#pragma unroll
        for (int ni = 0; ni < 4; ++ni) acc[mi][ni] = vzero8();

    const int arow  = tid >> 1;
    const int apart = (tid & 1) * 16;
    const int bk    = tid >> 3;
    const int bn    = (tid & 7) * 16;

    for (int k0 = 0; k0 < K; k0 += 32) {
        // ---- stage A tile (128x32) into LDS as bf16 ----
        if (ABF16) {
            const int4* ga = (const int4*)((const unsigned short*)Av +
                              (size_t)(m0 + arow) * K + k0 + apart);
            *(int4*)&As[arow][apart]     = ga[0];
            *(int4*)&As[arow][apart + 8] = ga[1];
        } else {
            const float* ga = (const float*)Av + (size_t)(m0 + arow) * K + k0 + apart;
            alignas(16) float av[16];
#pragma unroll
            for (int i = 0; i < 4; ++i) *(float4*)(av + 4 * i) = ((const float4*)ga)[i];
            alignas(16) unsigned up[8];
#pragma unroll
            for (int i = 0; i < 8; ++i) up[i] = pk2bf(av[2 * i], av[2 * i + 1]);
            *(int4*)&As[arow][apart]     = *(int4*)up;
            *(int4*)&As[arow][apart + 8] = *(int4*)(up + 4);
        }
        // ---- stage B tile (32x128) into LDS transposed [n][k] as bf16 ----
        {
            const float* gb = B + (size_t)(k0 + bk) * N + n0 + bn;
            alignas(16) float bv[16];
#pragma unroll
            for (int i = 0; i < 4; ++i) *(float4*)(bv + 4 * i) = ((const float4*)gb)[i];
#pragma unroll
            for (int i = 0; i < 16; ++i) Bs[bn + i][bk] = f2bf(bv[i]);
        }
        if (k0 + 32 < K) {
            __builtin_prefetch((const char*)B + ((size_t)(k0 + 32 + bk) * N + n0 + bn) * 4, 0, 1);
            if (!ABF16)
                __builtin_prefetch((const char*)Av + ((size_t)(m0 + arow) * K + k0 + 32 + apart) * 4, 0, 1);
        }
        __syncthreads();

        // ---- load ALL fragments first (12 x ds_load_b128), then WMMA burst ----
        v16bf afr[2], bfr[4];
        {
            const int m  = lane & 15;
            const int ka = (lane < 16) ? 0 : 8;
#pragma unroll
            for (int mi = 0; mi < 2; ++mi) {
                const unsigned short* r = &As[wm + mi * 16 + m][0];
                afr[mi] = frag2(r + ka, r + ka + 16);
            }
            const int n  = lane & 15;
            const int kb = (lane < 16) ? 0 : 8;
#pragma unroll
            for (int ni = 0; ni < 4; ++ni) {
                const unsigned short* r = &Bs[wn + ni * 16 + n][0];
                bfr[ni] = frag2(r + kb, r + kb + 16);
            }
        }
#pragma unroll
        for (int ni = 0; ni < 4; ++ni)
#pragma unroll
            for (int mi = 0; mi < 2; ++mi)
                acc[mi][ni] = __builtin_amdgcn_wmma_f32_16x16x32_bf16(
                    false, afr[mi], false, bfr[ni], (short)0, acc[mi][ni], false, false);
        __syncthreads();
    }

    // ---- writeout (C layout: lane<16 -> M=0..7, lane>=16 -> M=8..15, N=lane%16) ----
    const int cn = lane & 15;
    const int cm = (lane >> 4) * 8;
#pragma unroll
    for (int mi = 0; mi < 2; ++mi)
#pragma unroll
        for (int ni = 0; ni < 4; ++ni)
#pragma unroll
            for (int j = 0; j < 8; ++j) {
                int row = m0 + wm + mi * 16 + cm + j;
                int col = n0 + wn + ni * 16 + cn;
                C[(size_t)row * N + col] = acc[mi][ni][j];
            }
}

// ======================================================================
// Kernel 2/4: per-(t, head) RMSNorm + RoPE, pack q/k/v to bf16 workspace.
// grid = (2048, 48): y<32 -> q head, 32<=y<40 -> k head, y>=40 -> v head.
// ======================================================================
__global__ __launch_bounds__(128) void rmsnorm_rope_pack(
    const int* __restrict__ pos, const float* __restrict__ qkv,
    const float* __restrict__ qn, const float* __restrict__ kn,
    unsigned short* __restrict__ q, unsigned short* __restrict__ k,
    unsigned short* __restrict__ v)
{
    const int t   = blockIdx.x;
    const int hid = blockIdx.y;
    const int d   = threadIdx.x;

    const float* src;
    if (hid < 32)      src = qkv + (size_t)t * QKV_OUT + hid * HEAD_DIM;
    else if (hid < 40) src = qkv + (size_t)t * QKV_OUT + Q_SIZE + (hid - 32) * HEAD_DIM;
    else               src = qkv + (size_t)t * QKV_OUT + Q_SIZE + KV_SIZE + (hid - 40) * HEAD_DIM;

    const float x = src[d];
    if (hid >= 40) {  // v: plain bf16 pack (block-uniform branch)
        v[((size_t)t * NUM_KV + (hid - 40)) * HEAD_DIM + d] = f2bf(x);
        return;
    }

    // RMS over 128 elements (4 waves): DPP 16-lane + xor16 shuffle + LDS combine
    __shared__ float red[4];
    float ss = dpp_add16(x * x);
    ss += __shfl_xor(ss, 16, 32);
    if ((threadIdx.x & 31) == 0) red[threadIdx.x >> 5] = ss;
    __syncthreads();
    const float tot   = red[0] + red[1] + red[2] + red[3];
    const float scale = rsqrtf(tot * (1.0f / HEAD_DIM) + 1e-6f);

    const float* wn = (hid < 32) ? qn : kn;
    const float xn = x * scale * wn[d];
    const int   p  = d ^ 64;
    const float xp = src[p] * scale * wn[p];

    const int   i   = d & 63;
    const float fr  = powf(1.0e6f, -(float)i * (1.0f / 64.0f));
    const float ang = (float)pos[t] * fr;
    const float c = cosf(ang), s = sinf(ang);
    const float o = (d < 64) ? (xn * c - xp * s) : (xn * c + xp * s);

    if (hid < 32) q[((size_t)t * NUM_HEADS + hid) * HEAD_DIM + d] = f2bf(o);
    else          k[((size_t)t * NUM_KV + (hid - 32)) * HEAD_DIM + d] = f2bf(o);
}

// ======================================================================
// Kernel 3/4: causal flash attention (GQA, G=4) with bf16 WMMA.
// grid = (16 q-tiles, 32 heads), 256 thr; wave w owns 16 q-rows.
// K tile staged by the Tensor Data Mover (row-major == B-frag friendly);
// V tile staged TRANSPOSED by threads so the PV B-frags are 2 b128 loads.
// Softmax row reductions use DPP16 butterflies (no LDS traffic).
// ======================================================================
struct AttnSmem {
    unsigned short K[32][128];   // kv-row major: B-frag for Q.K^T reads rows
    unsigned short Vt[128][32];  // transposed V: [d][kv], B-frag for P.V
    unsigned short P[8][16][32]; // per-wave prob bounce (C-layout -> A-layout)
};

__global__ __launch_bounds__(256) void attn_flash_wmma(
    const unsigned short* __restrict__ qw, const unsigned short* __restrict__ kw,
    const unsigned short* __restrict__ vw, unsigned short* __restrict__ ow)
{
    __shared__ AttnSmem sm;
    const int tid  = threadIdx.x;
    const int lane = tid & 31;
    const int w    = tid >> 5;
    const int h    = blockIdx.y;
    const int kh   = h >> 2;
    const int q0   = blockIdx.x * 128;
    const int qr   = q0 + w * 16;          // this wave's first q row

    // Q fragments for 16x128 rows (4 chunks of K=32), loaded direct from global
    v16bf qf[4];
    {
        const int m  = lane & 15;
        const int kb = (lane < 16) ? 0 : 8;
        const unsigned short* qrow = qw + (size_t)(qr + m) * (NUM_HEADS * HEAD_DIM) + h * HEAD_DIM;
#pragma unroll
        for (int c = 0; c < 4; ++c) {
            ((int4*)&qf[c])[0] = *(const int4*)(qrow + c * 32 + kb);
            ((int4*)&qf[c])[1] = *(const int4*)(qrow + c * 32 + kb + 16);
        }
    }

    v8f   o[8];
#pragma unroll
    for (int dt = 0; dt < 8; ++dt) o[dt] = vzero8();
    float rmax[8], rsum[8];
#pragma unroll
    for (int j = 0; j < 8; ++j) { rmax[j] = -3.0e38f; rsum[j] = 0.0f; }

    const int cn = lane & 15;
    const int cm = (lane >> 4) * 8;

    // V-transpose staging indices: thread -> (kv row, 16 d columns)
    const int vkv = tid >> 3;
    const int vd  = (tid & 7) * 16;

    for (int s0 = 0; s0 < q0 + 128; s0 += 32) {
        // ---- TDM: stage K 32x128 bf16 tile into LDS (wave 0 issues) ----
        if (w == 0) {
            tdm_load_tile_2d((unsigned)offsetof(AttnSmem, K),
                             kw + ((size_t)s0 * NUM_KV + kh) * HEAD_DIM,
                             32u, 128u, (unsigned)(NUM_KV * HEAD_DIM));
        }
        // ---- all threads: stage V tile transposed [d][kv] ----
        {
            const unsigned short* gv =
                vw + ((size_t)(s0 + vkv) * NUM_KV + kh) * HEAD_DIM + vd;
            alignas(16) unsigned short vv[16];
            *(int4*)vv       = ((const int4*)gv)[0];
            *(int4*)(vv + 8) = ((const int4*)gv)[1];
#pragma unroll
            for (int i = 0; i < 16; ++i) sm.Vt[vd + i][vkv] = vv[i];
        }
        if (w == 0) __builtin_amdgcn_s_wait_tensorcnt(0);
        __syncthreads();

        // ---- scores S = Q . K^T : load all 8 B-frags, then 8-WMMA burst ----
        v8f sc0 = vzero8(), sc1 = vzero8();
        {
            const int kb = (lane < 16) ? 0 : 8;
            v16bf b0[4], b1[4];
#pragma unroll
            for (int c = 0; c < 4; ++c) {
                const unsigned short* r0 = &sm.K[cn][c * 32 + kb];
                const unsigned short* r1 = &sm.K[cn + 16][c * 32 + kb];
                b0[c] = frag2(r0, r0 + 16);
                b1[c] = frag2(r1, r1 + 16);
            }
#pragma unroll
            for (int c = 0; c < 4; ++c) {
                sc0 = __builtin_amdgcn_wmma_f32_16x16x32_bf16(false, qf[c], false, b0[c], (short)0, sc0, false, false);
                sc1 = __builtin_amdgcn_wmma_f32_16x16x32_bf16(false, qf[c], false, b1[c], (short)0, sc1, false, false);
            }
        }

        // ---- online softmax over the 32 new columns (DPP reductions) ----
        float alpha[8];
#pragma unroll
        for (int j = 0; j < 8; ++j) {
            const int tq = qr + cm + j;
            float v0 = sc0[j] * ATTN_SCALE;
            float v1 = sc1[j] * ATTN_SCALE;
            if (s0 + cn > tq)      v0 = -1.0e30f;
            if (s0 + 16 + cn > tq) v1 = -1.0e30f;
            const float t  = dpp_max16(fmaxf(v0, v1));
            const float nm = fmaxf(rmax[j], t);
            alpha[j] = __expf(rmax[j] - nm);
            const float p0 = __expf(v0 - nm);
            const float p1 = __expf(v1 - nm);
            const float ps = dpp_add16(p0 + p1);
            rsum[j] = rsum[j] * alpha[j] + ps;
            rmax[j] = nm;
            sm.P[w][cm + j][cn]      = f2bf(p0);
            sm.P[w][cm + j][16 + cn] = f2bf(p1);
        }
#pragma unroll
        for (int dt = 0; dt < 8; ++dt)
#pragma unroll
            for (int j = 0; j < 8; ++j) o[dt][j] *= alpha[j];

        // ---- P (A-frag, 16x32) from wave-private LDS bounce ----
        v16bf pf;
        {
            const int m  = lane & 15;
            const int ka = (lane < 16) ? 0 : 8;
            const unsigned short* r = &sm.P[w][m][0];
            pf = frag2(r + ka, r + ka + 16);
        }
        // ---- O += P . V : batches of 4 B-frags + 4 WMMAs ----
        {
            const int kb = (lane < 16) ? 0 : 8;
#pragma unroll
            for (int half = 0; half < 2; ++half) {
                v16bf bv[4];
#pragma unroll
                for (int i = 0; i < 4; ++i) {
                    const unsigned short* r = &sm.Vt[(half * 4 + i) * 16 + cn][kb];
                    bv[i] = frag2(r, r + 16);
                }
#pragma unroll
                for (int i = 0; i < 4; ++i)
                    o[half * 4 + i] = __builtin_amdgcn_wmma_f32_16x16x32_bf16(
                        false, pf, false, bv[i], (short)0, o[half * 4 + i], false, false);
            }
        }
        __syncthreads();
    }

    // ---- normalize and store bf16 attention output [T, 32*128] ----
#pragma unroll
    for (int dt = 0; dt < 8; ++dt)
#pragma unroll
        for (int j = 0; j < 8; ++j) {
            const int row = qr + cm + j;
            const int col = h * HEAD_DIM + dt * 16 + cn;
            ow[(size_t)row * (NUM_HEADS * HEAD_DIM) + col] = f2bf(o[dt][j] / rsum[j]);
        }
}

// ======================================================================
// Launcher
// ======================================================================
extern "C" void kernel_launch(void* const* d_in, const int* in_sizes, int n_in,
                              void* d_out, int out_size, void* d_ws, size_t ws_size,
                              hipStream_t stream) {
    (void)in_sizes; (void)n_in; (void)out_size; (void)ws_size;
    const int*   positions = (const int*)d_in[0];
    const float* hidden    = (const float*)d_in[1];
    const float* w_qkv     = (const float*)d_in[2];
    const float* q_norm    = (const float*)d_in[3];
    const float* k_norm    = (const float*)d_in[4];
    const float* w_o       = (const float*)d_in[5];
    float*       out       = (float*)d_out;

    char* ws = (char*)d_ws;
    float*          qkv  = (float*)(ws);                                  // 48 MiB
    unsigned short* qws  = (unsigned short*)(ws + 50331648);              // 16 MiB
    unsigned short* kws  = (unsigned short*)(ws + 50331648 + 16777216);   //  4 MiB
    unsigned short* vws  = (unsigned short*)(ws + 50331648 + 20971520);   //  4 MiB
    unsigned short* aws  = (unsigned short*)(ws + 50331648 + 25165824);   // 16 MiB

    // 1) qkv = hidden @ w_qkv   [2048,4096]x[4096,6144]
    gemm_bf16_wmma<false><<<dim3(QKV_OUT / 128, T_SEQ / 128), 256, 0, stream>>>(
        hidden, w_qkv, qkv, T_SEQ, QKV_OUT, HIDDEN);
    // 2) rmsnorm + rope + bf16 pack
    rmsnorm_rope_pack<<<dim3(T_SEQ, 48), 128, 0, stream>>>(
        positions, qkv, q_norm, k_norm, qws, kws, vws);
    // 3) causal flash attention
    attn_flash_wmma<<<dim3(T_SEQ / 128, NUM_HEADS), 256, 0, stream>>>(
        qws, kws, vws, aws);
    // 4) out = attn @ w_o   [2048,4096]x[4096,4096]
    gemm_bf16_wmma<true><<<dim3(HIDDEN / 128, T_SEQ / 128), 256, 0, stream>>>(
        aws, w_o, out, T_SEQ, HIDDEN, HIDDEN);
}
